// SpatioTemporalGCN_44908178047709
// MI455X (gfx1250) — matrix operand order repaired and checked by reading, not compiled, not run on hardware
//
#include <hip/hip_runtime.h>
#include <hip/hip_bf16.h>

typedef __attribute__((ext_vector_type(16))) __bf16 v16bf;
typedef __attribute__((ext_vector_type(8)))  __bf16 v8bf;
typedef __attribute__((ext_vector_type(8)))  float  v8f;
typedef __attribute__((ext_vector_type(4)))  float  v4f;

#define NEG_SLOPE 0.01f

// ---------- f32 -> bf16 (RNE) ----------
static __device__ __forceinline__ __bf16 f2bf(float f) {
  unsigned u = __builtin_bit_cast(unsigned, f);
  unsigned r = u + 0x7FFFu + ((u >> 16) & 1u);
  unsigned short h = (unsigned short)(r >> 16);
  return __builtin_bit_cast(__bf16, h);
}

// A fragment (16x32 bf16, MxK): lane<16 -> row M=lane, K {0..15};
// lanes 16..31 -> K {16..31}. p = &A[row][kstep*32] (bf16). Two b128 loads.
static __device__ __forceinline__ v16bf load_a_bf(const __bf16* __restrict__ p, int half) {
  const __bf16* q = p + half * 16;
  v8bf lo = *(const v8bf*)(q);
  v8bf hi = *(const v8bf*)(q + 8);
  return __builtin_shufflevector(lo, hi, 0, 1, 2, 3, 4, 5, 6, 7,
                                 8, 9, 10, 11, 12, 13, 14, 15);
}

// B fragment (32x16 bf16, KxN) from a contiguous B^T row: lane holds column
// N=lane%16, K = half*16 .. +15 contiguous. Two b128 loads.
static __device__ __forceinline__ v16bf load_bt_bf(const __bf16* __restrict__ p, int half) {
  const __bf16* q = p + half * 16;
  v8bf lo = *(const v8bf*)(q);
  v8bf hi = *(const v8bf*)(q + 8);
  return __builtin_shufflevector(lo, hi, 0, 1, 2, 3, 4, 5, 6, 7,
                                 8, 9, 10, 11, 12, 13, 14, 15);
}

// Same B^T-row fragment but from f32 memory: 4x b128 nontemporal loads + convert.
static __device__ __forceinline__ v16bf load_bt_f32_nt(const float* __restrict__ p, int half) {
  const v4f* q = (const v4f*)(p + half * 16);
  v4f a = __builtin_nontemporal_load(q);
  v4f b = __builtin_nontemporal_load(q + 1);
  v4f c = __builtin_nontemporal_load(q + 2);
  v4f d = __builtin_nontemporal_load(q + 3);
  v16bf o;
#pragma unroll
  for (int i = 0; i < 4; ++i) {
    o[i] = f2bf(a[i]); o[4 + i] = f2bf(b[i]);
    o[8 + i] = f2bf(c[i]); o[12 + i] = f2bf(d[i]);
  }
  return o;
}

static __device__ __forceinline__ v8f wmma_bf16(v16bf a, v16bf b, v8f c) {
  return __builtin_amdgcn_wmma_f32_16x16x32_bf16(false, a, false, b, (short)0, c, false, false);
}

static __device__ __forceinline__ float sigf(float x) { return 1.0f / (1.0f + __expf(-x)); }

// ---------------------------------------------------------------
__global__ void k_zero(float* __restrict__ p, long n) {
  long i = (long)blockIdx.x * blockDim.x + threadIdx.x;
  if (i < n) p[i] = 0.0f;
}

// packed f32 -> bf16 conversion, 8 elems/thread, b128 in / b128 out
__global__ void k_f32_to_bf16(const float* __restrict__ src, __bf16* __restrict__ dst, long n) {
  long i = ((long)blockIdx.x * blockDim.x + threadIdx.x) * 8;
  if (i >= n) return;
  v4f a = *(const v4f*)(src + i);
  v4f b = *(const v4f*)(src + i + 4);
  v8bf o;
#pragma unroll
  for (int j = 0; j < 4; ++j) { o[j] = f2bf(a[j]); o[4 + j] = f2bf(b[j]); }
  *(v8bf*)(dst + i) = o;
}

// W_gcn [3][64][64] (k-major) -> bf16 transposed [3][n][k] so B^T rows are contiguous
__global__ void k_wgcn_transpose(const float* __restrict__ W, __bf16* __restrict__ Wt) {
  int i = blockIdx.x * blockDim.x + threadIdx.x;  // 3*4096
  int l = i >> 12, rem = i & 4095;
  int n = rem >> 6, k = rem & 63;
  Wt[l * 4096 + n * 64 + k] = f2bf(W[l * 4096 + k * 64 + n]);
}

__global__ void k_deg_init(float* __restrict__ deg, int N) {
  int i = blockIdx.x * blockDim.x + threadIdx.x;
  if (i < N) deg[i] = 1.0f;  // self loop
}

__global__ void k_deg_count(const int* __restrict__ col, float* __restrict__ deg, int E) {
  int e = blockIdx.x * blockDim.x + threadIdx.x;
  if (e < E) atomicAdd(&deg[col[e]], 1.0f);
}

__global__ void k_deg_rsqrt(float* __restrict__ deg, int N) {
  int i = blockIdx.x * blockDim.x + threadIdx.x;
  if (i < N) deg[i] = rsqrtf(deg[i]);  // deg >= 1 always
}

// t = h @ W  (N x 64 @ 64 x 64), bf16 in, f32 out. One wave = 16 rows x 64 cols.
__global__ void k_gcn_gemm(const __bf16* __restrict__ h, const __bf16* __restrict__ Wt,
                           float* __restrict__ t, int N) {
  const int wave = blockIdx.x * (blockDim.x >> 5) + (threadIdx.x >> 5);
  const int lane = threadIdx.x & 31;
  const int half = lane >> 4, r = lane & 15;
  const long row0 = (long)wave * 16;
  if (row0 >= N) return;  // wave-uniform: EXEC stays all-ones for WMMA
  const __bf16* arow = h + (row0 + r) * 64;
  v8f acc[4];
#pragma unroll
  for (int i = 0; i < 4; ++i) acc[i] = (v8f){0, 0, 0, 0, 0, 0, 0, 0};
#pragma unroll
  for (int ks = 0; ks < 2; ++ks) {
    v16bf a = load_a_bf(arow + ks * 32, half);
#pragma unroll
    for (int nt = 0; nt < 4; ++nt) {
      v16bf b = load_bt_bf(Wt + (nt * 16 + r) * 64 + ks * 32, half);
      acc[nt] = wmma_bf16(a, b, acc[nt]);
    }
  }
#pragma unroll
  for (int nt = 0; nt < 4; ++nt)
#pragma unroll
    for (int v = 0; v < 8; ++v)
      t[(row0 + half * 8 + v) * 64 + nt * 16 + r] = acc[nt][v];
}

// acc[col] += dinv[row] * t[row]; one wave per edge, float2 per lane (coalesced 256B)
__global__ void k_gcn_scatter(const int* __restrict__ row, const int* __restrict__ col,
                              const float* __restrict__ dinv, const float* __restrict__ t,
                              float* __restrict__ acc, int E) {
  const int wv = blockIdx.x * (blockDim.x >> 5) + (threadIdx.x >> 5);
  if (wv >= E) return;
  const int lane = threadIdx.x & 31;
  const int rs = __builtin_amdgcn_readfirstlane(row[wv]);
  const int cd = __builtin_amdgcn_readfirstlane(col[wv]);
  const float dr = dinv[rs];
  const float2 v = ((const float2*)(t + (long)rs * 64))[lane];
  float* ac = acc + (long)cd * 64 + lane * 2;
  atomicAdd(ac, dr * v.x);
  atomicAdd(ac + 1, dr * v.y);
}

// h_new = LeakyReLU(dinv[n]*(acc + dinv[n]*t) + b) (+ x residual on last layer), stored bf16
__global__ void k_gcn_finalize(const float* __restrict__ t, const float* __restrict__ acc,
                               const float* __restrict__ dinv, const float* __restrict__ bias,
                               const float* __restrict__ resid, __bf16* __restrict__ hout,
                               long NC) {
  long i = (long)blockIdx.x * blockDim.x + threadIdx.x;
  if (i >= NC) return;
  long n = i >> 6;
  int c = (int)(i & 63);
  float d = dinv[n];
  float v = d * (acc[i] + d * t[i]) + bias[c];
  v = (v >= 0.0f) ? v : NEG_SLOPE * v;
  if (resid != nullptr) v += resid[i];
  hout[i] = f2bf(v);
}

// xg[256,256] += seq[256,K] @ W_ih^T ; grid=(16 m-tiles, 64 k-chunks), 1 wave/block.
// All operands bf16; wave keeps 16 n-tile accumulators to reuse each A fragment 16x.
__global__ void k_lstm_in_gemm(const __bf16* __restrict__ seq, const __bf16* __restrict__ Wih,
                               float* __restrict__ xg, int K, int NCOL) {
  const int mtile = blockIdx.x;
  const int kchunk = K / gridDim.y;
  const long k0 = (long)blockIdx.y * kchunk;
  const int lane = threadIdx.x & 31;
  const int half = lane >> 4, r = lane & 15;
  const __bf16* arow = seq + (long)(mtile * 16 + r) * K + k0;
  v8f acc[16];
#pragma unroll
  for (int i = 0; i < 16; ++i) acc[i] = (v8f){0, 0, 0, 0, 0, 0, 0, 0};
  for (int k = 0; k < kchunk; k += 32) {
    __builtin_prefetch(arow + k + 1024, 0, 1);  // global_prefetch: stream activations
    v16bf a = load_a_bf(arow + k, half);
#pragma unroll
    for (int nt = 0; nt < 16; ++nt) {
      const __bf16* brow = Wih + (long)(nt * 16 + r) * K + k0 + k;  // B^T row = W_ih row
      v16bf b = load_bt_bf(brow, half);
      acc[nt] = wmma_bf16(a, b, acc[nt]);
    }
  }
#pragma unroll
  for (int nt = 0; nt < 16; ++nt)
#pragma unroll
    for (int v = 0; v < 8; ++v) {
      int m = mtile * 16 + half * 8 + v;
      int n = nt * 16 + r;
      atomicAdd(&xg[(long)m * NCOL + n], acc[nt][v]);
    }
}

// 16-step LSTM recurrence, single block (B=16, H=64, 4H=256, T=16).
__global__ void k_lstm_steps(const float* __restrict__ xg, const float* __restrict__ Whh,
                             const float* __restrict__ bih, const float* __restrict__ bhh,
                             float* __restrict__ hn_out) {
  __shared__ float sh[16 * 64];
  __shared__ float sc[16 * 64];
  __shared__ float sg[16 * 256];
  const int tid = threadIdx.x;  // 256
  for (int i = tid; i < 1024; i += 256) { sh[i] = 0.0f; sc[i] = 0.0f; }
  __syncthreads();
  const int j = tid;  // gate unit 0..255
  const float bj = bih[j] + bhh[j];
  for (int t = 0; t < 16; ++t) {
    for (int b = 0; b < 16; ++b) {
      float d = 0.0f;
#pragma unroll 8
      for (int k = 0; k < 64; ++k) d += sh[b * 64 + k] * Whh[j * 64 + k];
      sg[b * 256 + j] = xg[((long)b * 16 + t) * 256 + j] + bj + d;  // seq row = b*T + t
    }
    __syncthreads();
    for (int i = tid; i < 1024; i += 256) {
      int b = i >> 6, u = i & 63;
      float gi = sg[b * 256 + u];
      float gf = sg[b * 256 + 64 + u];
      float gg = sg[b * 256 + 128 + u];
      float go = sg[b * 256 + 192 + u];
      float c = sigf(gf) * sc[i] + sigf(gi) * tanhf(gg);
      sc[i] = c;
      sh[i] = sigf(go) * tanhf(c);
    }
    __syncthreads();
  }
  for (int i = tid; i < 1024; i += 256) hn_out[i] = sh[i];
}

// z1 = relu(hn @ W1^T + b1); z2 = relu(z1 @ W2^T + b2); z2 stored bf16 for out GEMM
__global__ void k_mlp12(const float* __restrict__ hn, const float* __restrict__ W1,
                        const float* __restrict__ b1, const float* __restrict__ W2,
                        const float* __restrict__ b2, __bf16* __restrict__ z2out) {
  __shared__ float shn[16 * 64];
  __shared__ float sz1[16 * 128];
  const int tid = threadIdx.x;  // 256
  for (int i = tid; i < 1024; i += 256) shn[i] = hn[i];
  __syncthreads();
  for (int i = tid; i < 2048; i += 256) {
    int b = i >> 7, jj = i & 127;
    float d = b1[jj];
#pragma unroll 8
    for (int k = 0; k < 64; ++k) d += shn[b * 64 + k] * W1[jj * 64 + k];
    sz1[i] = fmaxf(d, 0.0f);
  }
  __syncthreads();
  for (int i = tid; i < 4096; i += 256) {
    int b = i >> 8, jj = i & 255;
    float d = b2[jj];
#pragma unroll 8
    for (int k = 0; k < 128; ++k) d += sz1[b * 128 + k] * W2[jj * 128 + k];
    z2out[i] = f2bf(fmaxf(d, 0.0f));
  }
}

// out[16, DIN] = z2[16,256] @ W3^T + b3 ; one wave per 16x16 n-tile, K=256 (8 k-steps).
// W3 is single-use: nontemporal b128 f32 loads, converted in-register.
__global__ void k_out_gemm(const __bf16* __restrict__ z2, const float* __restrict__ W3,
                           const float* __restrict__ b3, float* __restrict__ out, int DIN) {
  const int wave = blockIdx.x * (blockDim.x >> 5) + (threadIdx.x >> 5);  // n-tile
  const int lane = threadIdx.x & 31;
  const int half = lane >> 4, r = lane & 15;
  const int n = wave * 16 + r;
  const __bf16* arow = z2 + r * 256;
  const float* brow = W3 + (long)n * 256;  // B^T row = W3 row (contiguous)
  v8f acc = (v8f){0, 0, 0, 0, 0, 0, 0, 0};
#pragma unroll
  for (int ks = 0; ks < 8; ++ks) {
    v16bf a = load_a_bf(arow + ks * 32, half);
    v16bf b = load_bt_f32_nt(brow + ks * 32, half);
    acc = wmma_bf16(a, b, acc);
  }
  const float bias = b3[n];
#pragma unroll
  for (int v = 0; v < 8; ++v)
    out[(long)(half * 8 + v) * DIN + n] = acc[v] + bias;
}

// ---------------------------------------------------------------
extern "C" void kernel_launch(void* const* d_in, const int* in_sizes, int n_in,
                              void* d_out, int out_size, void* d_ws, size_t ws_size,
                              hipStream_t stream) {
  const float* x   = (const float*)d_in[0];
  const int*   ei  = (const int*)d_in[1];
  const float* Wg  = (const float*)d_in[2];
  const float* bg  = (const float*)d_in[3];
  const float* Wih = (const float*)d_in[4];
  const float* Whh = (const float*)d_in[5];
  const float* bih = (const float*)d_in[6];
  const float* bhh = (const float*)d_in[7];
  const float* W1  = (const float*)d_in[8];
  const float* b1  = (const float*)d_in[9];
  const float* W2  = (const float*)d_in[10];
  const float* b2  = (const float*)d_in[11];
  const float* W3  = (const float*)d_in[12];
  const float* b3  = (const float*)d_in[13];

  const int  C    = 64;                 // from W_gcn [3,64,64]
  const long NC   = in_sizes[0];        // 16777216
  const int  N    = (int)(NC / C);      // 262144 nodes
  const int  E    = in_sizes[1] / 2;    // 2097152 edges
  const int  DIN  = in_sizes[13];       // 65536
  const int  G4   = in_sizes[6];        // 256 = 4*HID
  const int  BT   = (int)(NC / DIN);    // 256 = B*T (B=T=16 in this problem)
  const long NWIH = in_sizes[4];        // 16777216

  // workspace layout (float units)
  float* bufB = (float*)d_ws;           // t = h @ W            (NC f32)
  float* bufC = bufB + NC;              // scatter accumulator  (NC f32)
  float* dinv = bufC + NC;              // rsqrt(degree)        (N f32)
  float* xg   = dinv + N;               // LSTM input gates     (BT*G4 f32)
  float* hn   = xg + (long)BT * G4;     // final hidden         (1024 f32)
  __bf16* hbf   = (__bf16*)(hn + 1024);         // node features, bf16 (NC)
  __bf16* Wihbf = hbf + NC;                     // W_ih bf16           (NWIH)
  __bf16* Wgtbf = Wihbf + NWIH;                 // W_gcn^T bf16        (3*C*C)
  __bf16* z2bf  = Wgtbf + 3 * C * C;            // MLP intermediate    (4096)

  const dim3 b256(256);

  // one-time (per launch) conversions: x -> bf16, W_ih -> bf16, W_gcn -> bf16^T
  k_f32_to_bf16<<<(int)(NC / 8 / 256), b256, 0, stream>>>(x, hbf, NC);
  k_f32_to_bf16<<<(int)(NWIH / 8 / 256), b256, 0, stream>>>(Wih, Wihbf, NWIH);
  k_wgcn_transpose<<<(3 * C * C) / 256, b256, 0, stream>>>(Wg, Wgtbf);

  // degree + normalization
  k_deg_init<<<(N + 255) / 256, b256, 0, stream>>>(dinv, N);
  k_deg_count<<<(E + 255) / 256, b256, 0, stream>>>(ei + E, dinv, E);
  k_deg_rsqrt<<<(N + 255) / 256, b256, 0, stream>>>(dinv, N);

  // 3 GCN layers (hbf is both input and output of each layer; kernels serialize on stream)
  for (int layer = 0; layer < 3; ++layer) {
    k_zero<<<(int)((NC + 255) / 256), b256, 0, stream>>>(bufC, NC);
    k_gcn_gemm<<<N / (16 * 8), b256, 0, stream>>>(hbf, Wgtbf + layer * C * C, bufB, N);
    k_gcn_scatter<<<(E + 7) / 8, b256, 0, stream>>>(ei, ei + E, dinv, bufB, bufC, E);
    k_gcn_finalize<<<(int)((NC + 255) / 256), b256, 0, stream>>>(
        bufB, bufC, dinv, bg + layer * C, (layer == 2) ? x : nullptr, hbf, NC);
  }

  // LSTM input GEMM (seq = hbf reshaped [256, 65536], bf16)
  k_zero<<<(BT * G4 + 255) / 256, b256, 0, stream>>>(xg, (long)BT * G4);
  k_lstm_in_gemm<<<dim3(BT / 16, 64), dim3(32), 0, stream>>>(hbf, Wihbf, xg, DIN, G4);

  // recurrence + MLP head
  k_lstm_steps<<<1, b256, 0, stream>>>(xg, Whh, bih, bhh, hn);
  k_mlp12<<<1, b256, 0, stream>>>(hn, W1, b1, W2, b2, z2bf);
  k_out_gemm<<<DIN / (16 * 8), b256, 0, stream>>>(z2bf, W3, b3, (float*)d_out, DIN);
}
